// PointerNet_58025008169238
// MI455X (gfx1250) — matrix operand order repaired and checked by reading, not compiled, not run on hardware
//
#include <hip/hip_runtime.h>
#include <hip/hip_bf16.h>
#include <stdint.h>

// ---------------------------------------------------------------------------
// PointerNet forward for MI455X (gfx1250, wave32, WMMA bf16 16x16x32).
// B=64, S=256, H=512, 4H=2048.
// ---------------------------------------------------------------------------

#define BDIM 64
#define SDIM 256
#define HDIM 512
#define GDIM 2048            // 4*H
#define MROWS (BDIM * SDIM)  // 16384

typedef __bf16 bf16_t;
typedef __bf16 bfv8  __attribute__((ext_vector_type(8)));
typedef __bf16 bfv16 __attribute__((ext_vector_type(16)));
typedef float  fv8   __attribute__((ext_vector_type(8)));

static __device__ inline bfv16 join8(bfv8 lo, bfv8 hi) {
  return __builtin_shufflevector(lo, hi, 0, 1, 2, 3, 4, 5, 6, 7,
                                 8, 9, 10, 11, 12, 13, 14, 15);
}

// A fragment (16x32 bf16, MxK): row-major A[M][lda], tile rows m0..m0+15.
// ISA layout: lanes 0-15 / 16-31 hold K chunks {0..7,16..23} / {8..15,24..31}.
// -> two contiguous 16-byte loads per lane.
static __device__ inline bfv16 frag_a(const bf16_t* A, int lda, int m0, int kb,
                                      int lane) {
  const bf16_t* p =
      A + (size_t)(m0 + (lane & 15)) * lda + kb + ((lane >> 4) << 3);
  bfv8 lo = *(const bfv8*)p;
  bfv8 hi = *(const bfv8*)(p + 16);
  return join8(lo, hi);
}

// B fragment (32x16 bf16, KxN): weights pre-transposed K-major:
// Wt[k*N + n] = W[n][k].  ISA layout: lane <-> N, lanes 0-15 / 16-31 hold
// K chunks {0..7,16..23} / {8..15,24..31} (same as A) -> strided scalar loads.
static __device__ inline bfv16 frag_b(const bf16_t* Wt, int N, int n0, int kb,
                                      int lane) {
  int n = n0 + (lane & 15);
  int kh = kb + ((lane >> 4) << 3);
  bfv16 r;
#pragma unroll
  for (int s = 0; s < 8; ++s) {
    r[s] = Wt[(size_t)(kh + s) * N + n];
    r[s + 8] = Wt[(size_t)(kh + 16 + s) * N + n];
  }
  return r;
}

static __device__ inline fv8 wmma_bf16(bfv16 a, bfv16 b, fv8 c) {
  return __builtin_amdgcn_wmma_f32_16x16x32_bf16(false, a, false, b, (short)0,
                                                 c, false, false);
}

static __device__ inline float sigf(float x) { return 1.f / (1.f + __expf(-x)); }

// ---------------------------------------------------------------------------
// Generic GEMM: C[M][N] = A[M][K] @ W[N][K]^T + bias, via K-major Wt.
// Block = 256 threads = 8 waves, wave computes one 16x16 tile; grid covers
// (N/128) x (M/16).
// ---------------------------------------------------------------------------
template <bool OUT_BF16>
__global__ void k_gemm(const bf16_t* __restrict__ A, int lda,
                       const bf16_t* __restrict__ Wt, int N,
                       const float* __restrict__ bias, void* __restrict__ Cout,
                       int ldc, int K) {
  int lane = threadIdx.x & 31;
  int wave = threadIdx.x >> 5;
  int n0 = (blockIdx.x * 8 + wave) * 16;
  int m0 = blockIdx.y * 16;
  fv8 acc = {};
  for (int kb = 0; kb < K; kb += 32) {
    bfv16 a = frag_a(A, lda, m0, kb, lane);
    bfv16 b = frag_b(Wt, N, n0, kb, lane);
    acc = wmma_bf16(a, b, acc);
  }
  int n = n0 + (lane & 15);
  int mb = m0 + ((lane >> 4) << 3);
  float bv = bias ? bias[n] : 0.f;
#pragma unroll
  for (int v = 0; v < 8; ++v) {
    float val = acc[v] + bv;
    size_t o = (size_t)(mb + v) * ldc + n;
    if (OUT_BF16)
      ((bf16_t*)Cout)[o] = (bf16_t)val;
    else
      ((float*)Cout)[o] = val;
  }
}

// ---------------------------------------------------------------------------
// Small prep kernels.
// ---------------------------------------------------------------------------
__global__ void k_cvt_tr(const float* __restrict__ W, bf16_t* __restrict__ Wt,
                         int N, int K) {  // Wt[k*N+n] = W[n*K+k]
  int i = blockIdx.x * blockDim.x + threadIdx.x;
  if (i >= N * K) return;
  int n = i / K, k = i % K;
  Wt[(size_t)k * N + n] = (bf16_t)W[(size_t)n * K + k];
}

__global__ void k_bias2(const float* a, const float* b, float* o, int n) {
  int i = blockIdx.x * blockDim.x + threadIdx.x;
  if (i < n) o[i] = a[i] + b[i];
}

// First embedding layer: K=10 (too small for WMMA) -> plain VALU.
__global__ void k_emb1(const float* __restrict__ x, const float* __restrict__ w1,
                       const float* __restrict__ b1, bf16_t* __restrict__ out) {
  int i = blockIdx.x * blockDim.x + threadIdx.x;
  if (i >= MROWS * 512) return;
  int r = i >> 9, j = i & 511;
  float s = b1[j];
#pragma unroll
  for (int k = 0; k < 10; ++k) s += x[r * 10 + k] * w1[j * 10 + k];
  out[i] = (bf16_t)s;
}

__global__ void k_rev_bf16(const bf16_t* __restrict__ in,
                           bf16_t* __restrict__ out,
                           const int* __restrict__ lens, int D) {
  int i = blockIdx.x * blockDim.x + threadIdx.x;
  if (i >= BDIM * SDIM * D) return;
  int d = i % D, t = (i / D) % SDIM, b = i / (D * SDIM);
  int len = lens[b];
  int st = (t < len) ? (len - 1 - t) : t;
  out[((size_t)b * SDIM + t) * D + d] = in[((size_t)b * SDIM + st) * D + d];
}

__global__ void k_rev_f32(const float* __restrict__ in, float* __restrict__ out,
                          const int* __restrict__ lens, int D) {
  int i = blockIdx.x * blockDim.x + threadIdx.x;
  if (i >= BDIM * SDIM * D) return;
  int d = i % D, t = (i / D) % SDIM, b = i / (D * SDIM);
  int len = lens[b];
  int st = (t < len) ? (len - 1 - t) : t;
  out[((size_t)b * SDIM + t) * D + d] = in[((size_t)b * SDIM + st) * D + d];
}

__global__ void k_concat(const float* __restrict__ f, const float* __restrict__ b,
                         bf16_t* __restrict__ out) {  // [16384][1024] bf16
  int i = blockIdx.x * blockDim.x + threadIdx.x;
  if (i >= MROWS * 1024) return;
  int row = i >> 10, d = i & 1023;
  float v = (d < 512) ? f[(size_t)row * 512 + d] : b[(size_t)row * 512 + d - 512];
  out[i] = (bf16_t)v;
}

__global__ void k_sum(const float* __restrict__ f, const float* __restrict__ b,
                      float* __restrict__ eo, bf16_t* __restrict__ ebf) {
  int i = blockIdx.x * blockDim.x + threadIdx.x;
  if (i >= MROWS * HDIM) return;
  float v = f[i] + b[i];
  eo[i] = v;
  ebf[i] = (bf16_t)v;
}

__global__ void k_mask_out(const int* __restrict__ lens, float* __restrict__ out) {
  int i = blockIdx.x * blockDim.x + threadIdx.x;
  if (i >= BDIM * SDIM * SDIM) return;
  int j = i & 255, t = (i >> 8) & 255, b = i >> 16;
  int len = lens[b];
  out[i] = (t < len && j < len) ? 1.f : 0.f;
}

__global__ void k_dec_init(const float* __restrict__ hfin,
                           const float* __restrict__ cfin,
                           bf16_t* __restrict__ dec_in, bf16_t* __restrict__ hb,
                           float* __restrict__ cdec) {
  int i = blockIdx.x * blockDim.x + threadIdx.x;
  if (i >= BDIM * HDIM) return;
  dec_in[i] = (bf16_t)0.f;
  hb[i] = (bf16_t)hfin[i];
  cdec[i] = cfin[i];
}

// ---------------------------------------------------------------------------
// Persistent one-workgroup LSTM over all 256 timesteps of one direction.
// h lives in LDS (bf16, A-fragments via ds_load_b128); Whh^T streams from L2.
// xproj[b*S+t][4H] already contains x@Wih^T + bih + bhh.
//
// Tile mapping: id = wave*16 + tt so m0 = (tt&3)*16 VARIES each iteration and
// `#pragma unroll 1` keeps the loop rolled -> no loop-invariant fragment
// hoisting, no scratch spills; per-iteration live set is ~24 VGPRs + acc.
// ---------------------------------------------------------------------------
__global__ void __launch_bounds__(1024) k_lstm_seq(
    const float* __restrict__ xproj, const bf16_t* __restrict__ WhhT,
    const int* __restrict__ lens, float* __restrict__ out,
    float* __restrict__ gates, float* __restrict__ cbuf,
    float* __restrict__ hfin, float* __restrict__ cfin) {
  extern __shared__ bf16_t hsm[];  // [64][512] bf16 = 64 KB
  int tid = threadIdx.x, lane = tid & 31, wave = tid >> 5;
  for (int i = tid; i < BDIM * HDIM; i += 1024) {
    hsm[i] = (bf16_t)0.f;
    cbuf[i] = 0.f;
  }
  __syncthreads();
  for (int t = 0; t < SDIM; ++t) {
    // gates = h @ Whh^T + xproj[:, t, :]   (C: 64x2048 -> 512 tiles / 32 waves)
#pragma unroll 1
    for (int tt = 0; tt < 16; ++tt) {
      int id = (wave << 4) + tt;   // m0 varies with tt -> no hoist/spill
      int m0 = (id & 3) << 4;
      int n0 = (id >> 2) << 4;
      fv8 acc = {};
#pragma unroll
      for (int kb = 0; kb < HDIM; kb += 32) {
        bfv16 a = frag_a(hsm, HDIM, m0, kb, lane);
        bfv16 b = frag_b(WhhT, GDIM, n0, kb, lane);
        acc = wmma_bf16(a, b, acc);
      }
      int n = n0 + (lane & 15);
      int mb = m0 + ((lane >> 4) << 3);
#pragma unroll
      for (int v = 0; v < 8; ++v) {
        int m = mb + v;  // batch index
        gates[(size_t)m * GDIM + n] =
            acc[v] + xproj[((size_t)m * SDIM + t) * GDIM + n];
      }
    }
    __threadfence();
    __syncthreads();
    // pointwise cell update; each thread owns 32 (b,j) pairs
    for (int r = 0; r < 32; ++r) {
      int idx = tid + (r << 10);
      int b = idx >> 9, j = idx & 511;
      bool valid = t < lens[b];
      const float* g = gates + (size_t)b * GDIM;
      float gi = g[j], gf = g[512 + j], gg = g[1024 + j], go = g[1536 + j];
      float c2 = sigf(gf) * cbuf[idx] + sigf(gi) * tanhf(gg);
      float h2 = sigf(go) * tanhf(c2);
      size_t orow = ((size_t)b * SDIM + t) * HDIM + j;
      if (valid) {
        cbuf[idx] = c2;
        hsm[idx] = (bf16_t)h2;
        out[orow] = h2;
      } else {
        out[orow] = 0.f;  // packed-sequence semantics: padded outputs are 0
      }
    }
    __syncthreads();
  }
  if (hfin) {
    for (int i = tid; i < BDIM * HDIM; i += 1024) {
      hfin[i] = (float)hsm[i];
      cfin[i] = cbuf[i];
    }
  }
}

// ---------------------------------------------------------------------------
// Decoder step: both LSTM cells (NOTE: cell2 uses the SAME old (h,c) as cell1,
// faithful to the reference) + W2h = h2 @ W2attn^T, all WMMA in one WG.
// Same anti-spill tile mapping as k_lstm_seq.
// ---------------------------------------------------------------------------
__global__ void __launch_bounds__(1024) k_dec_cells(
    const bf16_t* __restrict__ dec_in, bf16_t* __restrict__ hb,
    float* __restrict__ cdec, const bf16_t* __restrict__ W1ihT,
    const bf16_t* __restrict__ W1hhT, const float* __restrict__ bc1,
    const bf16_t* __restrict__ W2ihT, const bf16_t* __restrict__ W2hhT,
    const float* __restrict__ bc2, const bf16_t* __restrict__ WaT,
    float* __restrict__ g1, float* __restrict__ g2, float* __restrict__ w2h) {
  extern __shared__ bf16_t dsm[];  // h1 (64KB) + h2 (64KB)
  bf16_t* h1s = dsm;
  bf16_t* h2s = dsm + BDIM * HDIM;
  int tid = threadIdx.x, lane = tid & 31, wave = tid >> 5;

  // g1 = dec_in @ W1ih^T + h @ W1hh^T + bc1
#pragma unroll 1
  for (int tt = 0; tt < 16; ++tt) {
    int id = (wave << 4) + tt;
    int m0 = (id & 3) << 4, n0 = (id >> 2) << 4;
    fv8 acc = {};
#pragma unroll
    for (int kb = 0; kb < HDIM; kb += 32)
      acc = wmma_bf16(frag_a(dec_in, HDIM, m0, kb, lane),
                      frag_b(W1ihT, GDIM, n0, kb, lane), acc);
#pragma unroll
    for (int kb = 0; kb < HDIM; kb += 32)
      acc = wmma_bf16(frag_a(hb, HDIM, m0, kb, lane),
                      frag_b(W1hhT, GDIM, n0, kb, lane), acc);
    int n = n0 + (lane & 15), mb = m0 + ((lane >> 4) << 3);
#pragma unroll
    for (int v = 0; v < 8; ++v)
      g1[(size_t)(mb + v) * GDIM + n] = acc[v] + bc1[n];
  }
  __threadfence();
  __syncthreads();
  // h1 (c1 is discarded; cell2 re-reads old c)
  for (int r = 0; r < 32; ++r) {
    int idx = tid + (r << 10);
    int b = idx >> 9, j = idx & 511;
    const float* g = g1 + (size_t)b * GDIM;
    float c1 = sigf(g[512 + j]) * cdec[idx] + sigf(g[j]) * tanhf(g[1024 + j]);
    h1s[idx] = (bf16_t)(sigf(g[1536 + j]) * tanhf(c1));
  }
  __syncthreads();
  // g2 = h1 @ W2ih^T + h_old @ W2hh^T + bc2
#pragma unroll 1
  for (int tt = 0; tt < 16; ++tt) {
    int id = (wave << 4) + tt;
    int m0 = (id & 3) << 4, n0 = (id >> 2) << 4;
    fv8 acc = {};
#pragma unroll
    for (int kb = 0; kb < HDIM; kb += 32)
      acc = wmma_bf16(frag_a(h1s, HDIM, m0, kb, lane),
                      frag_b(W2ihT, GDIM, n0, kb, lane), acc);
#pragma unroll
    for (int kb = 0; kb < HDIM; kb += 32)
      acc = wmma_bf16(frag_a(hb, HDIM, m0, kb, lane),
                      frag_b(W2hhT, GDIM, n0, kb, lane), acc);
    int n = n0 + (lane & 15), mb = m0 + ((lane >> 4) << 3);
#pragma unroll
    for (int v = 0; v < 8; ++v)
      g2[(size_t)(mb + v) * GDIM + n] = acc[v] + bc2[n];
  }
  __threadfence();
  __syncthreads();
  // h2, c2 from OLD c
  for (int r = 0; r < 32; ++r) {
    int idx = tid + (r << 10);
    int b = idx >> 9, j = idx & 511;
    const float* g = g2 + (size_t)b * GDIM;
    float c2 = sigf(g[512 + j]) * cdec[idx] + sigf(g[j]) * tanhf(g[1024 + j]);
    float h2 = sigf(g[1536 + j]) * tanhf(c2);
    cdec[idx] = c2;
    h2s[idx] = (bf16_t)h2;
  }
  __syncthreads();
  // W2h = h2 @ W2attn^T (64x512, K=512): 128 tiles / 32 waves
#pragma unroll 1
  for (int i = 0; i < 4; ++i) {
    int id = (wave << 2) + i;  // m0 varies with i
    int m0 = (id & 3) << 4, n0 = (id >> 2) << 4;
    fv8 acc = {};
#pragma unroll
    for (int kb = 0; kb < HDIM; kb += 32)
      acc = wmma_bf16(frag_a(h2s, HDIM, m0, kb, lane),
                      frag_b(WaT, HDIM, n0, kb, lane), acc);
    int n = n0 + (lane & 15), mb = m0 + ((lane >> 4) << 3);
#pragma unroll
    for (int v = 0; v < 8; ++v) w2h[(size_t)(mb + v) * HDIM + n] = acc[v];
  }
  // publish h <- h2 for next step (last read of old hb was before the barrier)
  for (int i = tid; i < BDIM * HDIM; i += 1024) hb[i] = h2s[i];
}

// u[b][s] = vt . tanh(encT[b][s][:] + W2h[b][:]) ; one wave per s.
__global__ void k_attn_scores(const float* __restrict__ encT,
                              const float* __restrict__ w2h,
                              const float* __restrict__ vt,
                              float* __restrict__ u) {
  int b = blockIdx.y;
  int lane = threadIdx.x & 31, wave = threadIdx.x >> 5;
  int s = blockIdx.x * 8 + wave;
  const float* row = encT + ((size_t)b * SDIM + s) * HDIM;
  const float* hrow = w2h + (size_t)b * HDIM;
  float sum = 0.f;
  for (int k = lane; k < HDIM; k += 32) sum += vt[k] * tanhf(row[k] + hrow[k]);
#pragma unroll
  for (int off = 16; off > 0; off >>= 1) sum += __shfl_xor(sum, off, 32);
  if (lane == 0) u[b * SDIM + s] = sum;
}

// Masked log_softmax + argmax + gather of next decoder input. One block per b.
__global__ void k_attn_select(const float* __restrict__ u,
                              const int* __restrict__ lens, int t,
                              const float* __restrict__ enc_out,
                              float* __restrict__ logps,
                              float* __restrict__ idxs_out,
                              bf16_t* __restrict__ dec_in) {
  __shared__ float red[256];
  __shared__ int redi[256];
  __shared__ int sel;
  int b = blockIdx.x, s = threadIdx.x;
  int len = lens[b];
  bool m = (t < len) && (s < len);
  float v = u[b * SDIM + s] + (m ? 0.f : -103.6163f);  // log(1e-45)
  red[s] = v;
  __syncthreads();
  for (int o = 128; o > 0; o >>= 1) {
    if (s < o) red[s] = fmaxf(red[s], red[s + o]);
    __syncthreads();
  }
  float mx = red[0];
  __syncthreads();
  red[s] = __expf(v - mx);
  __syncthreads();
  for (int o = 128; o > 0; o >>= 1) {
    if (s < o) red[s] += red[s + o];
    __syncthreads();
  }
  float lse = logf(red[0]) + mx;
  __syncthreads();
  float logp = v - lse;
  logps[((size_t)b * SDIM + t) * SDIM + s] = logp;
  red[s] = m ? logp : -1e7f;
  redi[s] = s;
  __syncthreads();
  for (int o = 128; o > 0; o >>= 1) {
    if (s < o) {
      if (red[s + o] > red[s] ||
          (red[s + o] == red[s] && redi[s + o] < redi[s])) {
        red[s] = red[s + o];
        redi[s] = redi[s + o];
      }
    }
    __syncthreads();
  }
  if (s == 0) {
    sel = redi[0];
    idxs_out[b * SDIM + t] = (float)redi[0];
  }
  __syncthreads();
  int idx = sel;
  for (int k = s; k < HDIM; k += 256)
    dec_in[(size_t)b * HDIM + k] =
        (bf16_t)enc_out[((size_t)b * SDIM + idx) * HDIM + k];
}

// ---------------------------------------------------------------------------
// Host orchestration.
// ---------------------------------------------------------------------------
static inline int cdiv(int a, int b) { return (a + b - 1) / b; }

extern "C" void kernel_launch(void* const* d_in, const int* in_sizes, int n_in,
                              void* d_out, int out_size, void* d_ws,
                              size_t ws_size, hipStream_t stream) {
  (void)in_sizes; (void)n_in; (void)out_size; (void)ws_size;
  const float* input_seq = (const float*)d_in[0];
  const int* lens = (const int*)d_in[1];
  const float* w1 = (const float*)d_in[2];
  const float* b1 = (const float*)d_in[3];
  const float* w2 = (const float*)d_in[4];
  const float* b2 = (const float*)d_in[5];
  const float* w3 = (const float*)d_in[6];
  const float* b3 = (const float*)d_in[7];
  const float* W1a = (const float*)d_in[48];
  const float* W2a = (const float*)d_in[49];
  const float* vt = (const float*)d_in[50];

  // LSTM dirs: enc1L0 f/b, enc1L1 f/b, enc2L0 f/b, enc2L1 f/b, dec1, dec2
  const int baseIdx[10] = {8, 12, 16, 20, 24, 28, 32, 36, 40, 44};
  const int dirK[10] = {256, 256, 1024, 1024, 1024, 1024, 1024, 1024, 512, 512};

  size_t off = 0;
  auto alloc = [&](size_t bytes) -> void* {
    void* p = (char*)d_ws + off;
    off += (bytes + 255) & ~(size_t)255;
    return p;
  };

  bf16_t* w2t = (bf16_t*)alloc((size_t)512 * 512 * 2);
  bf16_t* w3t = (bf16_t*)alloc((size_t)512 * 256 * 2);
  bf16_t* W1t = (bf16_t*)alloc((size_t)512 * 512 * 2);
  bf16_t* W2t = (bf16_t*)alloc((size_t)512 * 512 * 2);
  bf16_t* wihT[10];
  bf16_t* whhT[10];
  float* bc[10];
  for (int d = 0; d < 10; ++d) {
    wihT[d] = (bf16_t*)alloc((size_t)dirK[d] * GDIM * 2);
    whhT[d] = (bf16_t*)alloc((size_t)512 * GDIM * 2);
    bc[d] = (float*)alloc((size_t)GDIM * 4);
  }
  bf16_t* e1 = (bf16_t*)alloc((size_t)MROWS * 512 * 2);
  bf16_t* e2 = (bf16_t*)alloc((size_t)MROWS * 512 * 2);
  bf16_t* Xe = (bf16_t*)alloc((size_t)MROWS * 256 * 2);
  bf16_t* X1 = (bf16_t*)alloc((size_t)MROWS * 1024 * 2);
  bf16_t* X2 = (bf16_t*)alloc((size_t)MROWS * 1024 * 2);
  bf16_t* XR = (bf16_t*)alloc((size_t)MROWS * 1024 * 2);
  float* xproj = (float*)alloc((size_t)MROWS * GDIM * 4);
  float* dir_f = (float*)alloc((size_t)MROWS * HDIM * 4);
  float* dir_b = (float*)alloc((size_t)MROWS * HDIM * 4);
  float* tmpo = (float*)alloc((size_t)MROWS * HDIM * 4);
  float* enc_out = (float*)alloc((size_t)MROWS * HDIM * 4);
  bf16_t* enc_bf = (bf16_t*)alloc((size_t)MROWS * HDIM * 2);
  float* encT = (float*)alloc((size_t)MROWS * HDIM * 4);
  float* gates = (float*)alloc((size_t)BDIM * GDIM * 4);
  float* g2buf = (float*)alloc((size_t)BDIM * GDIM * 4);
  float* cstate = (float*)alloc((size_t)BDIM * HDIM * 4);
  float* hfin = (float*)alloc((size_t)BDIM * HDIM * 4);
  float* cfin = (float*)alloc((size_t)BDIM * HDIM * 4);
  float* cdec = (float*)alloc((size_t)BDIM * HDIM * 4);
  bf16_t* hb = (bf16_t*)alloc((size_t)BDIM * HDIM * 2);
  bf16_t* dec_in = (bf16_t*)alloc((size_t)BDIM * HDIM * 2);
  float* w2h = (float*)alloc((size_t)BDIM * HDIM * 4);
  float* ubuf = (float*)alloc((size_t)BDIM * SDIM * 4);

  auto gemm_f32 = [&](const bf16_t* A, int lda, const bf16_t* Wt, int N,
                      const float* bias, float* C, int ldc, int K, int M) {
    k_gemm<false><<<dim3(N / 128, M / 16), 256, 0, stream>>>(A, lda, Wt, N,
                                                             bias, C, ldc, K);
  };
  auto gemm_bf = [&](const bf16_t* A, int lda, const bf16_t* Wt, int N,
                     const float* bias, bf16_t* C, int ldc, int K, int M) {
    k_gemm<true><<<dim3(N / 128, M / 16), 256, 0, stream>>>(A, lda, Wt, N,
                                                            bias, C, ldc, K);
  };

  // --- weight prep (fp32 -> bf16 K-major transpose; combined biases) ---
  k_cvt_tr<<<cdiv(512 * 512, 256), 256, 0, stream>>>(w2, w2t, 512, 512);
  k_cvt_tr<<<cdiv(256 * 512, 256), 256, 0, stream>>>(w3, w3t, 256, 512);
  k_cvt_tr<<<cdiv(512 * 512, 256), 256, 0, stream>>>(W1a, W1t, 512, 512);
  k_cvt_tr<<<cdiv(512 * 512, 256), 256, 0, stream>>>(W2a, W2t, 512, 512);
  for (int d = 0; d < 10; ++d) {
    const float* Wih = (const float*)d_in[baseIdx[d]];
    const float* Whh = (const float*)d_in[baseIdx[d] + 1];
    const float* bih = (const float*)d_in[baseIdx[d] + 2];
    const float* bhh = (const float*)d_in[baseIdx[d] + 3];
    k_cvt_tr<<<cdiv(GDIM * dirK[d], 256), 256, 0, stream>>>(Wih, wihT[d], GDIM,
                                                            dirK[d]);
    k_cvt_tr<<<cdiv(GDIM * 512, 256), 256, 0, stream>>>(Whh, whhT[d], GDIM,
                                                        512);
    k_bias2<<<cdiv(GDIM, 256), 256, 0, stream>>>(bih, bhh, bc[d], GDIM);
  }

  // --- embedding MLP ---
  k_emb1<<<cdiv(MROWS * 512, 256), 256, 0, stream>>>(input_seq, w1, b1, e1);
  gemm_bf(e1, 512, w2t, 512, b2, e2, 512, 512, MROWS);
  gemm_bf(e2, 512, w3t, 256, b3, Xe, 256, 512, MROWS);

  // --- 4 bidirectional LSTM layers ---
  bf16_t* Xcur = Xe;
  int curK = 256;
  bf16_t* pp[2] = {X1, X2};
  for (int L = 0; L < 4; ++L) {
    int fd = 2 * L, bd = 2 * L + 1;
    // forward direction
    gemm_f32(Xcur, curK, wihT[fd], GDIM, bc[fd], xproj, GDIM, curK, MROWS);
    k_lstm_seq<<<1, 1024, BDIM * HDIM * sizeof(bf16_t), stream>>>(
        xproj, whhT[fd], lens, dir_f, gates, cstate,
        (L == 3) ? hfin : nullptr, (L == 3) ? cfin : nullptr);
    // backward direction: reverse in, run forward, reverse out
    k_rev_bf16<<<cdiv(BDIM * SDIM * curK, 256), 256, 0, stream>>>(Xcur, XR,
                                                                  lens, curK);
    gemm_f32(XR, curK, wihT[bd], GDIM, bc[bd], xproj, GDIM, curK, MROWS);
    k_lstm_seq<<<1, 1024, BDIM * HDIM * sizeof(bf16_t), stream>>>(
        xproj, whhT[bd], lens, tmpo, gates, cstate, nullptr, nullptr);
    k_rev_f32<<<cdiv(BDIM * SDIM * HDIM, 256), 256, 0, stream>>>(tmpo, dir_b,
                                                                 lens, HDIM);
    if (L < 3) {
      k_concat<<<cdiv(MROWS * 1024, 256), 256, 0, stream>>>(dir_f, dir_b,
                                                            pp[L & 1]);
      Xcur = pp[L & 1];
      curK = 1024;
    }
  }
  // enc_out = fwd + bwd ; encT = enc_out @ W1^T
  k_sum<<<cdiv(MROWS * HDIM, 256), 256, 0, stream>>>(dir_f, dir_b, enc_out,
                                                     enc_bf);
  gemm_f32(enc_bf, HDIM, W1t, HDIM, nullptr, encT, HDIM, HDIM, MROWS);

  // --- outputs: mask tensor; decoder ---
  float* logps = (float*)d_out;
  float* idxs = logps + (size_t)BDIM * SDIM * SDIM;
  float* maskout = idxs + (size_t)BDIM * SDIM;
  k_mask_out<<<cdiv(BDIM * SDIM * SDIM, 256), 256, 0, stream>>>(lens, maskout);
  k_dec_init<<<cdiv(BDIM * HDIM, 256), 256, 0, stream>>>(hfin, cfin, dec_in,
                                                         hb, cdec);
  for (int t = 0; t < SDIM; ++t) {
    k_dec_cells<<<1, 1024, 2 * BDIM * HDIM * sizeof(bf16_t), stream>>>(
        dec_in, hb, cdec, wihT[8], whhT[8], bc[8], wihT[9], whhT[9], bc[9],
        W2t, gates, g2buf, w2h);
    k_attn_scores<<<dim3(SDIM / 8, BDIM), 256, 0, stream>>>(encT, w2h, vt,
                                                            ubuf);
    k_attn_select<<<BDIM, 256, 0, stream>>>(ubuf, lens, t, enc_out, logps,
                                            idxs, dec_in);
  }
}
